// TimeFrequencyTransformerNet_46342697124258
// MI455X (gfx1250) — compile-verified, matrix operand order checked by reading, not run on hardware
//
#include <hip/hip_runtime.h>
#include <math.h>

namespace {

constexpr int   kB = 2, kC = 4, kT = 1920, kE = 256, kNH = 8, kDh = 32;
constexpr int   kNFFT = 512, kHop = 128, kFH = 257, kFW = 16;
constexpr int   kSF = kFH * kFW;            // 4112
constexpr int   kSS = 2 * kSF;              // 8224
constexpr int   kWin = 512, kStride = 32;
constexpr int   kNwE = (kSS - kWin) / kStride + 1;   // 242
constexpr int   kNwD = (kT  - kWin) / kStride + 1;   // 45
constexpr int   kFrP = 128;                 // padded istft frames
constexpr float kInvSqrtDh = 0.17677669529663687f;   // 1/sqrt(32)
constexpr float kBnScale   = 0.9999950000374997f;    // 1/sqrt(1+1e-5)

typedef __attribute__((ext_vector_type(16))) _Float16 v16h;
typedef __attribute__((ext_vector_type(8)))  float    v8f;
typedef __attribute__((ext_vector_type(4)))  unsigned int u32x4;
typedef __attribute__((ext_vector_type(8)))  int          i32x8;
typedef __attribute__((ext_vector_type(4)))  int          i32x4;
union UH { v16h v; _Float16 h[16]; };
union UF { v8f  v; float    f[8];  };

#if defined(__has_builtin)
#if __has_builtin(__builtin_amdgcn_tensor_load_to_lds) && __has_builtin(__builtin_amdgcn_s_wait_tensorcnt)
#define GEMM_TDM 1
#endif
#endif
#ifndef GEMM_TDM
#define GEMM_TDM 0
#endif

// ---------------------------------------------------------------------------
// Generic strided/batched GEMM on the WMMA pipe.
//   C[z][m,n] = act(alpha * sum_k A[z][m,k]*B[z][k,n] + bias[n] (+ resid))
// A element: A + z*bA + m*sAm + k            (k-contiguous f32 -> f16 on load)
// B element: B + z*bB + k*sBk + n*sBn
// C element: C + z*bC + m*sCm + n
// One wave per block; wave computes a 32x16 C tile, K stepped by 64
// (4 v_wmma_f32_16x16x32_f16 per iteration).  A tile (32x64, zero-filled
// out-of-bounds, +1-dword row pad) is fetched by the Tensor Data Mover into
// LDS; B tile is loaded cooperatively (branch-free clamped loads).
// act: 0=none, 1=relu, 2=tanh.
// ---------------------------------------------------------------------------
__global__ __launch_bounds__(32)
void k_gemm(const float* __restrict__ A, const float* __restrict__ Bm,
            float* __restrict__ Cm, const float* __restrict__ bias,
            const float* __restrict__ resid,
            int M, int N, int K,
            long sAm, long bA,
            long sBk, long sBn, long bB,
            long sCm, long bC,
            float alpha, int act)
{
  __shared__ float As[32][65];   // 65 = 64 + TDM pad dword, kills bank conflicts
  __shared__ float Bs[64][17];
  const int  lane = threadIdx.x;
  const int  n0 = blockIdx.x * 16;
  const int  m0 = blockIdx.y * 32;
  const long z  = blockIdx.z;
  A  += z * bA;  Bm += z * bB;  Cm += z * bC;
  const float* Rz = resid ? resid + z * bC : nullptr;

  v8f acc0 = {}, acc1 = {};
  for (int k0 = 0; k0 < K; k0 += 64) {
#if GEMM_TDM
    {
      // D# group0: count=1, lds_addr, global_addr, type=2 (ISA 8.3)
      unsigned lds = (unsigned)(unsigned long long)(void*)&As[0][0];
      unsigned long long ga =
          (unsigned long long)(const void*)(A + (long)m0 * sAm + k0);
      int td0 = K - k0;              // remaining K: OOB elements read as zero
      int td1 = M - m0;              // remaining rows: OOB rows read as zero
      u32x4 g0;
      g0[0] = 1u;
      g0[1] = lds;
      g0[2] = (unsigned)(ga & 0xffffffffull);
      g0[3] = (unsigned)((ga >> 32) & 0x01ffffffull) | (2u << 30);
      // D# group1 (ISA 8.4): data_size=4B, pad_enable, pad every 64 dwords by
      // one dword; tile 64x32; tensor dims = remaining K x remaining M.
      i32x8 g1;
      g1[0] = (2 << 16) | (1 << 20) | (5 << 22);
      g1[1] = (td0 & 0xffff) << 16;                       // tensor_dim0[15:0]
      g1[2] = ((td0 >> 16) & 0xffff) | ((td1 & 0xffff) << 16);
      g1[3] = ((td1 >> 16) & 0xffff) | (64 << 16);        // tile_dim0 = 64
      g1[4] = 32;                                         // tile_dim1 = 32
      g1[5] = (int)(sAm & 0xffffffffl);                   // dim0 stride lo32
      g1[6] = (int)((sAm >> 32) & 0xffffl);               // dim0 stride hi16
      g1[7] = 0;
      i32x4 gz4 = {0, 0, 0, 0};                           // groups 2/3 unused
      i32x8 gz8 = {0, 0, 0, 0, 0, 0, 0, 0};               // unused 6th-arg group
      __builtin_amdgcn_tensor_load_to_lds(g0, g1, gz4, gz4, gz8, 0);
    }
#else
    for (int idx = lane; idx < 32 * 64; idx += 32) {      // A tile fallback
      int mm = idx >> 6, kk = idx & 63;
      int mi = m0 + mm, ki = k0 + kk;
      int mc = mi < M - 1 ? mi : M - 1;
      int kc = ki < K - 1 ? ki : K - 1;
      float v = A[(long)mc * sAm + kc];                   // always in-bounds
      As[mm][kk] = (mi < M && ki < K) ? v : 0.0f;
    }
#endif
    for (int idx = lane; idx < 64 * 16; idx += 32) {      // B tile 64x16
      int kk = idx >> 4, nn = idx & 15;
      int ki = k0 + kk, ni = n0 + nn;
      int kc = ki < K - 1 ? ki : K - 1;
      int nc = ni < N - 1 ? ni : N - 1;
      float v = Bm[(long)kc * sBk + (long)nc * sBn];      // always in-bounds
      Bs[kk][nn] = (ki < K && ni < N) ? v : 0.0f;
    }
    if (k0 + 64 < K)   // hint next B tile -> global_prefetch_b8
      __builtin_prefetch(Bm + (long)(k0 + 64) * sBk + (long)n0 * sBn, 0, 1);
#if GEMM_TDM
    __builtin_amdgcn_s_wait_tensorcnt(0);
#endif
    __syncthreads();

    const int half = lane >> 4, r16 = lane & 15;
#pragma unroll
    for (int ks = 0; ks < 64; ks += 32) {
      UH ua0, ua1, ub;
#pragma unroll
      for (int v = 0; v < 8; ++v) {
        // f16 A 16x32 layout: lanes 0-15 hold K 0..7,16..23; lanes 16-31 hold
        // K 8..15,24..31 (2 packed per VGPR).  B mirrored with N across lanes.
        int kk = ks + ((v < 4) ? 0 : 16) + half * 8 + 2 * (v & 3);
        ua0.h[2 * v]     = (_Float16)As[r16][kk];
        ua0.h[2 * v + 1] = (_Float16)As[r16][kk + 1];
        ua1.h[2 * v]     = (_Float16)As[16 + r16][kk];
        ua1.h[2 * v + 1] = (_Float16)As[16 + r16][kk + 1];
        ub.h[2 * v]      = (_Float16)Bs[kk][r16];
        ub.h[2 * v + 1]  = (_Float16)Bs[kk + 1][r16];
      }
      acc0 = __builtin_amdgcn_wmma_f32_16x16x32_f16(false, ua0.v, false, ub.v,
                                                    (short)0, acc0, false, false);
      acc1 = __builtin_amdgcn_wmma_f32_16x16x32_f16(false, ua1.v, false, ub.v,
                                                    (short)0, acc1, false, false);
    }
    __syncthreads();
  }

  UF u0, u1; u0.v = acc0; u1.v = acc1;
  const int nn = lane & 15;
  const int mb = (lane >> 4) * 8;      // C layout: lanes 16-31 -> rows +8
  if ((n0 + nn) < N) {
    float bv = bias ? bias[n0 + nn] : 0.0f;
#pragma unroll
    for (int r = 0; r < 8; ++r) {
#pragma unroll
      for (int t = 0; t < 2; ++t) {
        int mm = m0 + t * 16 + mb + r;
        if (mm < M) {
          float v = alpha * (t ? u1.f[r] : u0.f[r]) + bv;
          if (Rz) v += Rz[(long)mm * sCm + (n0 + nn)];
          if (act == 1) v = v > 0.0f ? v : 0.0f;
          else if (act == 2) v = tanhf(v);
          Cm[(long)mm * sCm + (n0 + nn)] = v;
        }
      }
    }
  }
}

// ------------------------------- reductions --------------------------------
__global__ __launch_bounds__(256)
void k_softmax(float* __restrict__ x, long cols)
{
  float* p = x + (long)blockIdx.x * cols;
  __shared__ float red[256];
  const int t = threadIdx.x;
  float mx = -3.0e38f;
  for (long c = t; c < cols; c += 256) mx = fmaxf(mx, p[c]);
  red[t] = mx; __syncthreads();
  for (int s = 128; s > 0; s >>= 1) { if (t < s) red[t] = fmaxf(red[t], red[t + s]); __syncthreads(); }
  mx = red[0]; __syncthreads();
  float sum = 0.0f;
  for (long c = t; c < cols; c += 256) { float e = __expf(p[c] - mx); p[c] = e; sum += e; }
  red[t] = sum; __syncthreads();
  for (int s = 128; s > 0; s >>= 1) { if (t < s) red[t] += red[t + s]; __syncthreads(); }
  float inv = 1.0f / red[0];
  for (long c = t; c < cols; c += 256) p[c] *= inv;
}

__global__ __launch_bounds__(256)
void k_ln(const float* __restrict__ x, const float* __restrict__ r,
          const float* __restrict__ w, const float* __restrict__ b,
          float* __restrict__ y)
{
  const long row = blockIdx.x;
  const int  t = threadIdx.x;
  float v = x[row * 256 + t] + (r ? r[row * 256 + t] : 0.0f);
  __shared__ float red[256];
  red[t] = v; __syncthreads();
  for (int s = 128; s > 0; s >>= 1) { if (t < s) red[t] += red[t + s]; __syncthreads(); }
  float mean = red[0] * (1.0f / 256.0f);
  __syncthreads();
  float d = v - mean;
  red[t] = d * d; __syncthreads();
  for (int s = 128; s > 0; s >>= 1) { if (t < s) red[t] += red[t + s]; __syncthreads(); }
  float var = red[0] * (1.0f / 256.0f);
  y[row * 256 + t] = d * rsqrtf(var + 1e-5f) * w[t] + b[t];
}

// ------------------------------- elementwise -------------------------------
__device__ inline float pe_val(int pos, int e) {
  int   e2  = e & ~1;
  float div = __expf((float)e2 * (-9.210340371976184f / 256.0f)); // -ln(1e4)/256
  float ang = (float)pos * div;
  return (e & 1) ? __cosf(ang) : __sinf(ang);
}
__device__ inline float hannw(int n) {
  return 0.5f * (1.0f - __cosf(6.283185307179586f * (float)n / 512.0f));
}

__global__ void k_zero(float* p, long n) {
  long i = (long)blockIdx.x * 256 + threadIdx.x; if (i < n) p[i] = 0.0f;
}

__global__ void k_im2col_time(const float* __restrict__ x, float* __restrict__ cols) {
  long i = (long)blockIdx.x * 256 + threadIdx.x;
  if (i >= (long)kB * kT * 28) return;
  int  c = (int)(i % 28); long row = i / 28;
  int  t = (int)(row % kT); int b = (int)(row / kT);
  int  ic = c / 7, dt = c % 7, j = t - 3 + dt;
  cols[i] = (j >= 0 && j < kT) ? x[((long)b * kC + ic) * kT + j] : 0.0f;
}

__global__ void k_post_time(const float* __restrict__ h, const float* __restrict__ bw,
                            const float* __restrict__ bb, float* __restrict__ out) {
  long i = (long)blockIdx.x * 256 + threadIdx.x;
  if (i >= (long)kB * kT * kE) return;
  int e = (int)(i % kE), t = (int)((i / kE) % kT);
  float v = h[i] * kBnScale * bw[e] + bb[e];
  out[i] = (v > 0.0f ? v : 0.0f) + pe_val(t, e);
}

__global__ void k_build_frames(const float* __restrict__ x, float* __restrict__ fr) {
  long i = (long)blockIdx.x * 256 + threadIdx.x;
  if (i >= (long)kB * kC * kFW * kNFFT) return;
  int n = (int)(i % kNFFT); long row = i / kNFFT;
  int f = (int)(row % kFW); long bc = row / kFW;
  int j = f * kHop + n - kNFFT / 2;                 // reflect pad
  if (j < 0) j = -j;
  if (j >= kT) j = 2 * (kT - 1) - j;
  fr[i] = x[bc * kT + j] * hannw(n);
}

__global__ void k_build_dft(float* __restrict__ d) {     // [512, 514]: cos | -sin
  long i = (long)blockIdx.x * 256 + threadIdx.x;
  if (i >= (long)kNFFT * 514) return;
  int col = (int)(i % 514), n = (int)(i / 514), k = col >> 1;
  float ang = 6.283185307179586f * (float)k * (float)n / 512.0f;
  d[i] = (col & 1) ? -__sinf(ang) : __cosf(ang);
}

__global__ void k_im2col_freq(const float* __restrict__ spec, float* __restrict__ cols, int ri) {
  long i = (long)blockIdx.x * 256 + threadIdx.x;
  if (i >= (long)kB * kFH * kFW * 36) return;
  int c = (int)(i % 36); long row = i / 36;
  int f = (int)(row % kFW); long t1 = row / kFW;
  int k = (int)(t1 % kFH), b = (int)(t1 / kFH);
  int ic = c / 9, r9 = c % 9, dh = r9 / 3, dw = r9 % 3;
  int hh = k - 1 + dh, ww = f - 1 + dw;
  float v = 0.0f;
  if (hh >= 0 && hh < kFH && ww >= 0 && ww < kFW)
    v = spec[(((long)b * kC + ic) * kFW + ww) * 514 + 2 * hh + ri];
  cols[i] = v;
}

__global__ void k_post_freq(const float* __restrict__ g, const float* __restrict__ bw,
                            const float* __restrict__ bb, float* __restrict__ out) {
  long i = (long)blockIdx.x * 256 + threadIdx.x;
  if (i >= (long)kB * kSF * kE) return;
  int  e = (int)(i % kE); long s = (i / kE) % kSF;
  int  k = (int)(s / kFW), f = (int)(s % kFW);
  float v = g[i] * kBnScale * bw[e] + bb[e];
  out[i] = (v > 0.0f ? v : 0.0f) + pe_val(k, e) + pe_val(f, e);
}

__global__ void k_interp(const float* __restrict__ tf, float* __restrict__ out) {
  long i = (long)blockIdx.x * 256 + threadIdx.x;
  if (i >= (long)kB * kSF * kE) return;
  int e = (int)(i % kE); long s = (i / kE) % kSF; int b = (int)(i / ((long)kSF * kE));
  float src = ((float)s + 0.5f) * ((float)kT / (float)kSF) - 0.5f;
  src = fminf(fmaxf(src, 0.0f), (float)(kT - 1));
  int i0 = (int)floorf(src);
  int i1 = (i0 + 1 < kT) ? i0 + 1 : kT - 1;
  float w = src - (float)i0;
  const float* tb = tf + (long)b * kT * kE;
  out[i] = tb[(long)i0 * kE + e] * (1.0f - w) + tb[(long)i1 * kE + e] * w;
}

// FiLM's seq-first quirk: attention mixes the batch axis (2x2) per (s, head).
__global__ void k_film_attn(const float* __restrict__ Q, const float* __restrict__ Kf,
                            const float* __restrict__ Vf, float* __restrict__ O) {
  long i = (long)blockIdx.x * 256 + threadIdx.x;
  if (i >= (long)kB * kSF * kNH) return;
  int h = (int)(i % kNH); long sh = i / kNH;
  int s = (int)(sh % kSF), bq = (int)(sh / kSF);
  const float* q  = Q  + ((long)bq * kSF + s) * kE + h * kDh;
  const float* k0 = Kf + ((long)0  * kSF + s) * kE + h * kDh;
  const float* k1 = Kf + ((long)1  * kSF + s) * kE + h * kDh;
  float d0 = 0.0f, d1 = 0.0f;
  for (int d = 0; d < kDh; ++d) { d0 += q[d] * k0[d]; d1 += q[d] * k1[d]; }
  d0 *= kInvSqrtDh; d1 *= kInvSqrtDh;
  float m = fmaxf(d0, d1);
  float e0 = __expf(d0 - m), e1 = __expf(d1 - m);
  float inv = 1.0f / (e0 + e1);
  float a0 = e0 * inv, a1 = e1 * inv;
  const float* v0 = Vf + ((long)0 * kSF + s) * kE + h * kDh;
  const float* v1 = Vf + ((long)1 * kSF + s) * kE + h * kDh;
  float* o = O + ((long)bq * kSF + s) * kE + h * kDh;
  for (int d = 0; d < kDh; ++d) o[d] = a0 * v0[d] + a1 * v1[d];
}

__global__ void k_modulate(const float* __restrict__ ff, const float* __restrict__ g,
                           const float* __restrict__ bt, float* __restrict__ out) {
  long i = (long)blockIdx.x * 256 + threadIdx.x;
  if (i < (long)kB * kSF * kE) out[i] = (1.0f + g[i]) * ff[i] + bt[i];
}

__global__ void k_interleave(const float* __restrict__ mr, const float* __restrict__ mi,
                             float* __restrict__ comb) {
  long i = (long)blockIdx.x * 256 + threadIdx.x;
  if (i >= (long)kB * kSS * kE) return;
  int e = (int)(i % kE); long s2 = (i / kE) % kSS; int b = (int)(i / ((long)kSS * kE));
  long s = s2 >> 1;
  long src = ((long)b * kSF + s) * kE + e;
  comb[i] = (s2 & 1) ? mi[src] : mr[src];
}

__global__ void k_copywin(const float* __restrict__ src, float* __restrict__ dst,
                          int s0, int Stot) {
  long i = (long)blockIdx.x * 256 + threadIdx.x;
  if (i >= (long)kB * kWin * kE) return;
  int b = (int)(i / ((long)kWin * kE)); long rem = i % ((long)kWin * kE);
  dst[i] = src[((long)b * Stot + s0) * kE + rem];
}

__global__ void k_scatter(float* __restrict__ acc, const float* __restrict__ src,
                          int s0, int Stot) {
  long i = (long)blockIdx.x * 256 + threadIdx.x;
  if (i >= (long)kB * kWin * kE) return;
  int b = (int)(i / ((long)kWin * kE)); long rem = i % ((long)kWin * kE);
  acc[((long)b * Stot + s0) * kE + rem] += src[i];
}

__global__ void k_wavg(const float* __restrict__ acc, float* __restrict__ out,
                       int Stot, int Nw) {
  long i = (long)blockIdx.x * 256 + threadIdx.x;
  if (i >= (long)kB * Stot * kE) return;
  int s = (int)((i / kE) % Stot);
  int wlo = (s >= kWin) ? (s - (kWin - kStride)) / kStride : 0;
  int whi = s / kStride; if (whi > Nw - 1) whi = Nw - 1;
  out[i] = acc[i] / (float)(whi - wlo + 1);
}

__global__ void k_build_spec_o(const float* __restrict__ fo, float* __restrict__ sp) {
  long i = (long)blockIdx.x * 256 + threadIdx.x;
  if (i >= (long)kB * kFrP * 514) return;
  int col = (int)(i % 514); long row = i / 514;
  int f = (int)(row % kFrP), b = (int)(row / kFrP);
  int k = col >> 1, ri = col & 1;
  float v = 0.0f;
  if (f < kFW) {                       // fr = fo[:,0::2,0], fi = fo[:,1::2,1]
    long s = (long)k * kFW + f;
    v = fo[((long)b * kSS + 2 * s + ri) * 2 + ri];
  }
  sp[i] = v;
}

__global__ void k_build_idft(float* __restrict__ d) {    // [514, 512]
  long i = (long)blockIdx.x * 256 + threadIdx.x;
  if (i >= (long)514 * kNFFT) return;
  int n = (int)(i % kNFFT), row = (int)(i / kNFFT);
  int k = row >> 1, ri = row & 1;
  float wk = (k == 0 || k == 256) ? 1.0f : 2.0f;
  float ang = 6.283185307179586f * (float)k * (float)n / 512.0f;
  d[i] = (ri ? -wk * __sinf(ang) : wk * __cosf(ang)) * (1.0f / 512.0f);
}

__global__ void k_ola(const float* __restrict__ fr, float* __restrict__ y) {
  long i = (long)blockIdx.x * 256 + threadIdx.x;
  if (i >= (long)kB * kT) return;
  int t = (int)(i % kT), b = (int)(i / kT);
  int l = t + kNFFT / 2;
  int flo = (l >= kNFFT) ? (l - (kNFFT - kHop)) / kHop : 0;
  int fhi = l / kHop; if (fhi > kFrP - 1) fhi = kFrP - 1;
  float num = 0.0f, den = 0.0f;
  for (int f = flo; f <= fhi; ++f) {
    int n = l - f * kHop;
    float wv = hannw(n);
    num += fr[((long)b * kFrP + f) * kNFFT + n] * wv;
    den += wv * wv;
  }
  y[i] = num / (den > 1e-11f ? den : 1.0f);
}

__global__ void k_final(const float* __restrict__ td, const float* __restrict__ yf,
                        const float* __restrict__ alphap, float* __restrict__ out) {
  long i = (long)blockIdx.x * 256 + threadIdx.x;
  if (i >= (long)kB * kT) return;
  float a = alphap[0];
  out[i] = a * td[i] + (1.0f - a) * yf[i];
}

// ------------------------------- host helpers ------------------------------
inline unsigned gb(long n) { return (unsigned)((n + 255) / 256); }

inline void gemm(hipStream_t st, const float* A, const float* Bm, float* C,
                 const float* bias, const float* resid,
                 int M, int N, int K,
                 long sAm, long bA,
                 long sBk, long sBn, long bB,
                 long sCm, long bC, int batch, float alpha, int act)
{
  dim3 g((unsigned)((N + 15) / 16), (unsigned)((M + 31) / 32), (unsigned)batch);
  k_gemm<<<g, 32, 0, st>>>(A, Bm, C, bias, resid, M, N, K,
                           sAm, bA, sBk, sBn, bB, sCm, bC, alpha, act);
}

} // anonymous namespace

extern "C" void kernel_launch(void* const* d_in, const int* in_sizes, int n_in,
                              void* d_out, int out_size, void* d_ws, size_t ws_size,
                              hipStream_t stream)
{
  (void)in_sizes; (void)n_in; (void)out_size; (void)ws_size;
  // setup_inputs() dict order: x, then params (insertion order, nested dicts
  // recursed in insertion order).
  const float* X          = (const float*)d_in[0];
  const float* conv1_w    = (const float*)d_in[1];
  const float* conv1_b    = (const float*)d_in[2];
  const float* bn1_w      = (const float*)d_in[3];
  const float* bn1_b      = (const float*)d_in[4];
  const float* conv2r_w   = (const float*)d_in[5];
  const float* conv2r_b   = (const float*)d_in[6];
  const float* bn2r_w     = (const float*)d_in[7];
  const float* bn2r_b     = (const float*)d_in[8];
  const float* conv2i_w   = (const float*)d_in[9];
  const float* conv2i_b   = (const float*)d_in[10];
  const float* bn2i_w     = (const float*)d_in[11];
  const float* bn2i_b     = (const float*)d_in[12];
  const float* film_in_w  = (const float*)d_in[13];
  const float* film_in_b  = (const float*)d_in[14];
  const float* film_out_w = (const float*)d_in[15];
  const float* film_out_b = (const float*)d_in[16];
  const float* filmn_w    = (const float*)d_in[17];
  const float* filmn_b    = (const float*)d_in[18];
  const float* gam_w      = (const float*)d_in[19];
  const float* gam_b      = (const float*)d_in[20];
  const float* bet_w      = (const float*)d_in[21];
  const float* bet_b      = (const float*)d_in[22];
  const float* e_in_w     = (const float*)d_in[23];
  const float* e_in_b     = (const float*)d_in[24];
  const float* e_out_w    = (const float*)d_in[25];
  const float* e_out_b    = (const float*)d_in[26];
  const float* e_ln1_w    = (const float*)d_in[27];
  const float* e_ln1_b    = (const float*)d_in[28];
  const float* e_ln2_w    = (const float*)d_in[29];
  const float* e_ln2_b    = (const float*)d_in[30];
  const float* e_l1_w     = (const float*)d_in[31];
  const float* e_l1_b     = (const float*)d_in[32];
  const float* e_l2_w     = (const float*)d_in[33];
  const float* e_l2_b     = (const float*)d_in[34];
  const float* d_sa_in_w  = (const float*)d_in[35];
  const float* d_sa_in_b  = (const float*)d_in[36];
  const float* d_sa_out_w = (const float*)d_in[37];
  const float* d_sa_out_b = (const float*)d_in[38];
  const float* d_ca_in_w  = (const float*)d_in[39];
  const float* d_ca_in_b  = (const float*)d_in[40];
  const float* d_ca_out_w = (const float*)d_in[41];
  const float* d_ca_out_b = (const float*)d_in[42];
  const float* d_ln1_w    = (const float*)d_in[43];
  const float* d_ln1_b    = (const float*)d_in[44];
  const float* d_ln2_w    = (const float*)d_in[45];
  const float* d_ln2_b    = (const float*)d_in[46];
  const float* d_ln3_w    = (const float*)d_in[47];
  const float* d_ln3_b    = (const float*)d_in[48];
  const float* d_l1_w     = (const float*)d_in[49];
  const float* d_l1_b     = (const float*)d_in[50];
  const float* d_l2_w     = (const float*)d_in[51];
  const float* d_l2_b     = (const float*)d_in[52];
  const float* fp_w       = (const float*)d_in[53];
  const float* fp_b       = (const float*)d_in[54];
  const float* tp_w       = (const float*)d_in[55];
  const float* tp_b       = (const float*)d_in[56];
  const float* alpha_p    = (const float*)d_in[57];
  float* OUT = (float*)d_out;

  float* base = (float*)d_ws;
  long   off  = 0;
  auto alloc = [&](long n) { float* p = base + off; off += n; return p; };

  float* timef  = alloc((long)kB * kT * kE);
  float* colsT  = alloc((long)kB * kT * 28);
  float* hT     = alloc((long)kB * kT * kE);
  float* frames = alloc((long)kB * kC * kFW * kNFFT);
  float* dftB   = alloc((long)kNFFT * 514);
  float* specRI = alloc((long)kB * kC * kFW * 514);
  float* colsF  = alloc((long)kB * kFH * kFW * 36);
  float* gF     = alloc((long)kB * kSF * kE);
  float* ffr    = alloc((long)kB * kSF * kE);
  float* ffi    = alloc((long)kB * kSF * kE);
  float* tfup   = alloc((long)kB * kSF * kE);
  float* Qf     = alloc((long)kB * kSF * kE);
  float* Kfb    = alloc((long)kB * kSF * kE);
  float* Vfb    = alloc((long)kB * kSF * kE);
  float* afilm  = alloc((long)kB * kSF * kE);
  float* aproj  = alloc((long)kB * kSF * kE);
  float* fmb    = alloc((long)kB * kSF * kE);
  float* gamb   = alloc((long)kB * kSF * kE);
  float* betb   = alloc((long)kB * kSF * kE);
  float* modr   = alloc((long)kB * kSF * kE);
  float* modi   = alloc((long)kB * kSF * kE);
  float* comb   = alloc((long)kB * kSS * kE);
  float* encAcc = alloc((long)kB * kSS * kE);
  float* encOut = alloc((long)kB * kSS * kE);
  float* xw     = alloc((long)kB * kWin * kE);
  float* Qe     = alloc((long)kB * kWin * kE);
  float* Ke     = alloc((long)kB * kWin * kE);
  float* Ve     = alloc((long)kB * kWin * kE);
  float* attn   = alloc((long)kB * kWin * kE);
  float* proj   = alloc((long)kB * kWin * kE);
  float* x1     = alloc((long)kB * kWin * kE);
  float* x2     = alloc((long)kB * kWin * kE);
  float* x3     = alloc((long)kB * kWin * kE);
  float* h1     = alloc((long)kB * kWin * 1024);
  float* h2     = alloc((long)kB * kWin * kE);
  float* sc     = alloc((long)kB * kNH * kWin * kWin);
  float* Kmem   = alloc((long)kB * kSS * kE);
  float* Vmem   = alloc((long)kB * kSS * kE);
  float* scd    = alloc((long)kB * kNH * kWin * kSS);
  float* Qc     = alloc((long)kB * kWin * kE);
  float* projc  = alloc((long)kB * kWin * kE);
  float* decAcc = alloc((long)kB * kT * kE);
  float* decOut = alloc((long)kB * kT * kE);
  float* fout   = alloc((long)kB * kSS * 2);
  float* specO  = alloc((long)kB * kFrP * 514);
  float* idftB  = alloc((long)514 * kNFFT);
  float* framesO= alloc((long)kB * kFrP * kNFFT);
  float* yfreq  = alloc((long)kB * kT);
  float* td     = alloc((long)kB * kT);

  // ---- time embedding: conv1d(k7,p3) via im2col GEMM, BN+ReLU+PE ----
  k_im2col_time<<<gb((long)kB * kT * 28), 256, 0, stream>>>(X, colsT);
  gemm(stream, colsT, conv1_w, hT, conv1_b, nullptr,
       kB * kT, kE, 28, 28, 0, 1, 28, 0, kE, 0, 1, 1.0f, 0);
  k_post_time<<<gb((long)kB * kT * kE), 256, 0, stream>>>(hT, bn1_w, bn1_b, timef);

  // ---- STFT as GEMM against DFT basis ----
  k_build_frames<<<gb((long)kB * kC * kFW * kNFFT), 256, 0, stream>>>(X, frames);
  k_build_dft<<<gb((long)kNFFT * 514), 256, 0, stream>>>(dftB);
  gemm(stream, frames, dftB, specRI, nullptr, nullptr,
       kB * kC * kFW, 514, kNFFT, kNFFT, 0, 514, 1, 0, 514, 0, 1, 1.0f, 0);

  // ---- freq embeddings (conv2d 3x3 via im2col GEMM) ----
  k_im2col_freq<<<gb((long)kB * kFH * kFW * 36), 256, 0, stream>>>(specRI, colsF, 0);
  gemm(stream, colsF, conv2r_w, gF, conv2r_b, nullptr,
       kB * kFH * kFW, kE, 36, 36, 0, 1, 36, 0, kE, 0, 1, 1.0f, 0);
  k_post_freq<<<gb((long)kB * kSF * kE), 256, 0, stream>>>(gF, bn2r_w, bn2r_b, ffr);
  k_im2col_freq<<<gb((long)kB * kFH * kFW * 36), 256, 0, stream>>>(specRI, colsF, 1);
  gemm(stream, colsF, conv2i_w, gF, conv2i_b, nullptr,
       kB * kFH * kFW, kE, 36, 36, 0, 1, 36, 0, kE, 0, 1, 1.0f, 0);
  k_post_freq<<<gb((long)kB * kSF * kE), 256, 0, stream>>>(gF, bn2i_w, bn2i_b, ffi);

  // ---- upsample time features, FiLM both branches ----
  k_interp<<<gb((long)kB * kSF * kE), 256, 0, stream>>>(timef, tfup);
  gemm(stream, tfup, film_in_w, Qf, film_in_b, nullptr,
       kB * kSF, kE, kE, kE, 0, 1, kE, 0, kE, 0, 1, 1.0f, 0);
  auto film = [&](const float* ff, float* mod) {
    gemm(stream, ff, film_in_w + 256 * 256, Kfb, film_in_b + 256, nullptr,
         kB * kSF, kE, kE, kE, 0, 1, kE, 0, kE, 0, 1, 1.0f, 0);
    gemm(stream, ff, film_in_w + 512 * 256, Vfb, film_in_b + 512, nullptr,
         kB * kSF, kE, kE, kE, 0, 1, kE, 0, kE, 0, 1, 1.0f, 0);
    k_film_attn<<<gb((long)kB * kSF * kNH), 256, 0, stream>>>(Qf, Kfb, Vfb, afilm);
    gemm(stream, afilm, film_out_w, aproj, film_out_b, nullptr,
         kB * kSF, kE, kE, kE, 0, 1, kE, 0, kE, 0, 1, 1.0f, 0);
    k_ln<<<dim3((unsigned)(kB * kSF)), 256, 0, stream>>>(tfup, aproj, filmn_w, filmn_b, fmb);
    gemm(stream, fmb, gam_w, gamb, gam_b, nullptr,
         kB * kSF, kE, kE, kE, 0, 1, kE, 0, kE, 0, 1, 1.0f, 2);   // tanh
    gemm(stream, fmb, bet_w, betb, bet_b, nullptr,
         kB * kSF, kE, kE, kE, 0, 1, kE, 0, kE, 0, 1, 1.0f, 0);
    k_modulate<<<gb((long)kB * kSF * kE), 256, 0, stream>>>(ff, gamb, betb, mod);
  };
  film(ffr, modr);
  film(ffi, modi);
  k_interleave<<<gb((long)kB * kSS * kE), 256, 0, stream>>>(modr, modi, comb);

  // ---- windowed transformer encoder (242 overlapping windows) ----
  k_zero<<<gb((long)kB * kSS * kE), 256, 0, stream>>>(encAcc, (long)kB * kSS * kE);
  for (int w = 0; w < kNwE; ++w) {
    const int s0 = w * kStride;
    k_copywin<<<gb((long)kB * kWin * kE), 256, 0, stream>>>(comb, xw, s0, kSS);
    gemm(stream, xw, e_in_w,             Qe, e_in_b,       nullptr, kWin, kE, kE,
         kE, (long)kWin * kE, 1, kE, 0, kE, (long)kWin * kE, kB, 1.0f, 0);
    gemm(stream, xw, e_in_w + 256 * 256, Ke, e_in_b + 256, nullptr, kWin, kE, kE,
         kE, (long)kWin * kE, 1, kE, 0, kE, (long)kWin * kE, kB, 1.0f, 0);
    gemm(stream, xw, e_in_w + 512 * 256, Ve, e_in_b + 512, nullptr, kWin, kE, kE,
         kE, (long)kWin * kE, 1, kE, 0, kE, (long)kWin * kE, kB, 1.0f, 0);
    for (int b = 0; b < kB; ++b)
      gemm(stream, Qe + (long)b * kWin * kE, Ke + (long)b * kWin * kE,
           sc + (long)b * kNH * kWin * kWin, nullptr, nullptr,
           kWin, kWin, kDh, kE, kDh, 1, kE, kDh,
           kWin, (long)kWin * kWin, kNH, kInvSqrtDh, 0);
    k_softmax<<<dim3((unsigned)(kB * kNH * kWin)), 256, 0, stream>>>(sc, kWin);
    for (int b = 0; b < kB; ++b)
      gemm(stream, sc + (long)b * kNH * kWin * kWin, Ve + (long)b * kWin * kE,
           attn + (long)b * kWin * kE, nullptr, nullptr,
           kWin, kDh, kWin, kWin, (long)kWin * kWin, kE, 1, kDh,
           kE, kDh, kNH, 1.0f, 0);
    gemm(stream, attn, e_out_w, proj, e_out_b, nullptr,
         kB * kWin, kE, kE, kE, 0, 1, kE, 0, kE, 0, 1, 1.0f, 0);
    k_ln<<<dim3((unsigned)(kB * kWin)), 256, 0, stream>>>(xw, proj, e_ln1_w, e_ln1_b, x1);
    gemm(stream, x1, e_l1_w, h1, e_l1_b, nullptr,
         kB * kWin, 1024, kE, kE, 0, 1, kE, 0, 1024, 0, 1, 1.0f, 1);
    gemm(stream, h1, e_l2_w, h2, e_l2_b, nullptr,
         kB * kWin, kE, 1024, 1024, 0, 1, 1024, 0, kE, 0, 1, 1.0f, 0);
    k_ln<<<dim3((unsigned)(kB * kWin)), 256, 0, stream>>>(x1, h2, e_ln2_w, e_ln2_b, x2);
    k_scatter<<<gb((long)kB * kWin * kE), 256, 0, stream>>>(encAcc, x2, s0, kSS);
  }
  k_wavg<<<gb((long)kB * kSS * kE), 256, 0, stream>>>(encAcc, encOut, kSS, kNwE);

  // ---- decoder: memory K/V once, then 45 windows ----
  gemm(stream, encOut, d_ca_in_w + 256 * 256, Kmem, d_ca_in_b + 256, nullptr,
       kB * kSS, kE, kE, kE, 0, 1, kE, 0, kE, 0, 1, 1.0f, 0);
  gemm(stream, encOut, d_ca_in_w + 512 * 256, Vmem, d_ca_in_b + 512, nullptr,
       kB * kSS, kE, kE, kE, 0, 1, kE, 0, kE, 0, 1, 1.0f, 0);
  k_zero<<<gb((long)kB * kT * kE), 256, 0, stream>>>(decAcc, (long)kB * kT * kE);
  for (int w = 0; w < kNwD; ++w) {
    const int s0 = w * kStride;
    k_copywin<<<gb((long)kB * kWin * kE), 256, 0, stream>>>(timef, xw, s0, kT);
    // self-attention
    gemm(stream, xw, d_sa_in_w,             Qe, d_sa_in_b,       nullptr, kWin, kE, kE,
         kE, (long)kWin * kE, 1, kE, 0, kE, (long)kWin * kE, kB, 1.0f, 0);
    gemm(stream, xw, d_sa_in_w + 256 * 256, Ke, d_sa_in_b + 256, nullptr, kWin, kE, kE,
         kE, (long)kWin * kE, 1, kE, 0, kE, (long)kWin * kE, kB, 1.0f, 0);
    gemm(stream, xw, d_sa_in_w + 512 * 256, Ve, d_sa_in_b + 512, nullptr, kWin, kE, kE,
         kE, (long)kWin * kE, 1, kE, 0, kE, (long)kWin * kE, kB, 1.0f, 0);
    for (int b = 0; b < kB; ++b)
      gemm(stream, Qe + (long)b * kWin * kE, Ke + (long)b * kWin * kE,
           sc + (long)b * kNH * kWin * kWin, nullptr, nullptr,
           kWin, kWin, kDh, kE, kDh, 1, kE, kDh,
           kWin, (long)kWin * kWin, kNH, kInvSqrtDh, 0);
    k_softmax<<<dim3((unsigned)(kB * kNH * kWin)), 256, 0, stream>>>(sc, kWin);
    for (int b = 0; b < kB; ++b)
      gemm(stream, sc + (long)b * kNH * kWin * kWin, Ve + (long)b * kWin * kE,
           attn + (long)b * kWin * kE, nullptr, nullptr,
           kWin, kDh, kWin, kWin, (long)kWin * kWin, kE, 1, kDh,
           kE, kDh, kNH, 1.0f, 0);
    gemm(stream, attn, d_sa_out_w, proj, d_sa_out_b, nullptr,
         kB * kWin, kE, kE, kE, 0, 1, kE, 0, kE, 0, 1, 1.0f, 0);
    k_ln<<<dim3((unsigned)(kB * kWin)), 256, 0, stream>>>(xw, proj, d_ln1_w, d_ln1_b, x1);
    // cross-attention to full memory (512 x 8224 scores per head)
    gemm(stream, x1, d_ca_in_w, Qc, d_ca_in_b, nullptr,
         kB * kWin, kE, kE, kE, 0, 1, kE, 0, kE, 0, 1, 1.0f, 0);
    for (int b = 0; b < kB; ++b)
      gemm(stream, Qc + (long)b * kWin * kE, Kmem + (long)b * kSS * kE,
           scd + (long)b * kNH * kWin * kSS, nullptr, nullptr,
           kWin, kSS, kDh, kE, kDh, 1, kE, kDh,
           kSS, (long)kWin * kSS, kNH, kInvSqrtDh, 0);
    k_softmax<<<dim3((unsigned)(kB * kNH * kWin)), 256, 0, stream>>>(scd, kSS);
    for (int b = 0; b < kB; ++b)
      gemm(stream, scd + (long)b * kNH * kWin * kSS, Vmem + (long)b * kSS * kE,
           attn + (long)b * kWin * kE, nullptr, nullptr,
           kWin, kDh, kSS, kSS, (long)kWin * kSS, kE, 1, kDh,
           kE, kDh, kNH, 1.0f, 0);
    gemm(stream, attn, d_ca_out_w, projc, d_ca_out_b, nullptr,
         kB * kWin, kE, kE, kE, 0, 1, kE, 0, kE, 0, 1, 1.0f, 0);
    k_ln<<<dim3((unsigned)(kB * kWin)), 256, 0, stream>>>(x1, projc, d_ln2_w, d_ln2_b, x2);
    gemm(stream, x2, d_l1_w, h1, d_l1_b, nullptr,
         kB * kWin, 1024, kE, kE, 0, 1, kE, 0, 1024, 0, 1, 1.0f, 1);
    gemm(stream, h1, d_l2_w, h2, d_l2_b, nullptr,
         kB * kWin, kE, 1024, 1024, 0, 1, 1024, 0, kE, 0, 1, 1.0f, 0);
    k_ln<<<dim3((unsigned)(kB * kWin)), 256, 0, stream>>>(x2, h2, d_ln3_w, d_ln3_b, x3);
    k_scatter<<<gb((long)kB * kWin * kE), 256, 0, stream>>>(decAcc, x3, s0, kT);
  }
  k_wavg<<<gb((long)kB * kT * kE), 256, 0, stream>>>(decAcc, decOut, kT, kNwD);

  // ---- output heads: freq branch (iSTFT via iDFT GEMM + OLA) + time branch ----
  gemm(stream, encOut, fp_w, fout, fp_b, nullptr,
       kB * kSS, 2, kE, kE, 0, 1, kE, 0, 2, 0, 1, 1.0f, 0);
  k_build_spec_o<<<gb((long)kB * kFrP * 514), 256, 0, stream>>>(fout, specO);
  k_build_idft<<<gb((long)514 * kNFFT), 256, 0, stream>>>(idftB);
  gemm(stream, specO, idftB, framesO, nullptr, nullptr,
       kB * kFrP, kNFFT, 514, 514, 0, kNFFT, 1, 0, kNFFT, 0, 1, 1.0f, 0);
  k_ola<<<gb((long)kB * kT), 256, 0, stream>>>(framesO, yfreq);
  gemm(stream, decOut, tp_w, td, tp_b, nullptr,
       kB * kT, 1, kE, kE, 0, 1, kE, 0, 1, 0, 1, 1.0f, 0);
  k_final<<<gb((long)kB * kT), 256, 0, stream>>>(td, yfreq, alpha_p, OUT);
}
